// Transformer_19816979104178
// MI455X (gfx1250) — matrix operand order repaired and checked by reading, not compiled, not run on hardware
//
#include <hip/hip_runtime.h>
#include <math.h>
#include <type_traits>

// ---------------- model dims ----------------
#define BB 4
#define SS 512
#define TT 512
#define DD 512
#define HH 8
#define LL 4
#define FF_ 2048
#define VOUT_ 32000

typedef __attribute__((ext_vector_type(16))) _Float16 v16h;
typedef __attribute__((ext_vector_type(8)))  float    v8f;

// ---------------- CDNA5 async copy to LDS ----------------
// GLOBAL_LOAD_ASYNC_TO_LDS_B128: per-lane 16B global->LDS, tracked by ASYNCcnt.
static __device__ __forceinline__ void async_copy_b128(unsigned lds_off, const void* gaddr)
{
    asm volatile("global_load_async_to_lds_b128 %0, %1, off"
                 :: "v"(lds_off), "v"(gaddr) : "memory");
}
static __device__ __forceinline__ void wait_asynccnt0()
{
    asm volatile("s_wait_asynccnt 0x0" ::: "memory");
}

// ---------------- WMMA GEMM ----------------
// C[z][m][n] = act( alpha * sum_k A[z][m][k] * B[z][k][n]  + bias[n] )
// TRANSB: B stored as [N][K] row-major, i.e. C = A * B^T.
// Block tile 128x64, outer K-step 64 (2 unrolled WMMA K=32 steps).
// 8 waves (4x2); each wave owns 32x32 (2x2 accumulators) -> 8 WMMAs/stage.
// f16 A and f16 TRANSB B are staged via async-to-LDS (f16 NN B by plain stores); fp32 sources convert while staging.
#define BM 128
#define BN 64
#define BKO 64
#define LDA_S 72   // halves; 144B rows -> 16B aligned
#define LDB_S 72

template<typename TA, typename TB, bool TRANSB, bool RELU, typename TC>
__global__ __launch_bounds__(256) void gemm_wmma(
    const TA* __restrict__ A, long baseA, long sAz, int lda,
    const TB* __restrict__ B, long baseB, long sBz, int ldb,
    TC* __restrict__ C, long baseC, long sCz, int ldc,
    int M, int N, int K, const float* __restrict__ bias, float alpha)
{
    constexpr bool A16 = std::is_same<TA, _Float16>::value;
    constexpr bool B16 = std::is_same<TB, _Float16>::value;

    __shared__ _Float16 As[BM * LDA_S];   // 18.4 KB
    __shared__ _Float16 Bs[BN * LDB_S];   //  9.2 KB

    const int tid  = threadIdx.x;
    const int wave = tid >> 5;
    const int lane = tid & 31;
    const int l15  = lane & 15;
    const int wm   = wave & 3;   // 0..3 -> 32-row slab
    const int wn   = wave >> 2;  // 0..1 -> 32-col slab

    const long z = blockIdx.z;
    const TA* Ag = A + baseA + z * sAz;
    const TB* Bg = B + baseB + z * sBz;
    TC*       Cg = C + baseC + z * sCz;

    const int gm0 = blockIdx.y * BM;
    const int gn0 = blockIdx.x * BN;
    (void)M;

    v8f acc00 = {}, acc01 = {}, acc10 = {}, acc11 = {};

    const int ar = tid >> 1;            // 0..127
    const int ac = (tid & 1) * 32;      // 0 or 32
    const int br = tid >> 2;            // 0..63
    const int bc = (tid & 3) * 16;      // 0,16,32,48

    const int koffA = (lane < 16) ? 0 : 16;
    const int koffB = (lane < 16) ? 0 : 16;

    for (int k0 = 0; k0 < K; k0 += BKO) {
        // ---- stage A tile (128x64) ----
        if constexpr (A16) {
            const _Float16* src = Ag + (long)(gm0 + ar) * lda + (k0 + ac);
            unsigned dls = (unsigned)(size_t)&As[ar * LDA_S + ac];
            #pragma unroll
            for (int c = 0; c < 4; ++c) async_copy_b128(dls + c * 16, src + c * 8);
        } else {
            const float* src = Ag + (long)(gm0 + ar) * lda + (k0 + ac);
            _Float16* dst = &As[ar * LDA_S + ac];
            #pragma unroll
            for (int j = 0; j < 32; ++j) dst[j] = (_Float16)src[j];
            if (k0 + BKO < K) __builtin_prefetch(src + BKO, 0, 0);
        }
        // ---- stage B tile -> LDS as [N][K] ----
        if constexpr (B16 && TRANSB) {
            const _Float16* src = Bg + (long)(gn0 + br) * ldb + (k0 + bc);
            unsigned dls = (unsigned)(size_t)&Bs[br * LDB_S + bc];
            async_copy_b128(dls, src);
            async_copy_b128(dls + 16, src + 8);
        } else if constexpr (B16) {
            const _Float16* src = Bg + (long)(k0 + br) * ldb + (gn0 + bc);
            #pragma unroll
            for (int j = 0; j < 16; ++j) Bs[(bc + j) * LDB_S + br] = src[j];
        } else if constexpr (TRANSB) {
            const float* src = Bg + (long)(gn0 + br) * ldb + (k0 + bc);
            _Float16* dst = &Bs[br * LDB_S + bc];
            #pragma unroll
            for (int j = 0; j < 16; ++j) dst[j] = (_Float16)src[j];
            if (k0 + BKO < K) __builtin_prefetch(src + BKO, 0, 0);
        } else {
            const float* src = Bg + (long)(k0 + br) * ldb + (gn0 + bc);
            #pragma unroll
            for (int j = 0; j < 16; ++j) Bs[(bc + j) * LDB_S + br] = (_Float16)src[j];
            if (k0 + BKO < K) __builtin_prefetch(src + (long)BKO * ldb, 0, 0);
        }
        if constexpr (A16 || (B16 && TRANSB)) wait_asynccnt0();
        __syncthreads();

        #pragma unroll
        for (int ki = 0; ki < BKO; ki += 32) {
            // A fragments (rows wm*32 + {0,16} + l15), contiguous layout:
            // per-lane K {off..off+15}
            const int ab0 = (wm * 32 + l15) * LDA_S + ki + koffA;
            const int ab1 = (wm * 32 + 16 + l15) * LDA_S + ki + koffA;
            v16h a0, a1;
            #pragma unroll
            for (int j = 0; j < 8; ++j) {
                a0[j] = As[ab0 + j];  a0[8 + j] = As[ab0 + 8 + j];
                a1[j] = As[ab1 + j];  a1[8 + j] = As[ab1 + 8 + j];
            }
            // B fragments (cols wn*32 + {0,16} + l15): contiguous K halves
            const int bb0 = (wn * 32 + l15) * LDB_S + ki + koffB;
            const int bb1 = (wn * 32 + 16 + l15) * LDB_S + ki + koffB;
            v16h b0, b1;
            #pragma unroll
            for (int j = 0; j < 16; ++j) { b0[j] = Bs[bb0 + j]; b1[j] = Bs[bb1 + j]; }

            acc00 = __builtin_amdgcn_wmma_f32_16x16x32_f16(false, a0, false, b0,
                                                           (short)0, acc00, false, false);
            acc01 = __builtin_amdgcn_wmma_f32_16x16x32_f16(false, a0, false, b1,
                                                           (short)0, acc01, false, false);
            acc10 = __builtin_amdgcn_wmma_f32_16x16x32_f16(false, a1, false, b0,
                                                           (short)0, acc10, false, false);
            acc11 = __builtin_amdgcn_wmma_f32_16x16x32_f16(false, a1, false, b1,
                                                           (short)0, acc11, false, false);
        }
        __syncthreads();
    }

    // ---- epilogue: C/D layout: elem j -> row j + 8*(lane>=16), col lane%16 ----
    const int rbase = gm0 + wm * 32 + ((lane < 16) ? 0 : 8);
    const int cbase = gn0 + wn * 32 + l15;
    v8f accs[4] = { acc00, acc01, acc10, acc11 };
    #pragma unroll
    for (int ti = 0; ti < 2; ++ti) {
        #pragma unroll
        for (int tj = 0; tj < 2; ++tj) {
            const v8f& a = accs[ti * 2 + tj];
            const int col = cbase + tj * 16;
            float bv = bias ? bias[col] : 0.0f;
            #pragma unroll
            for (int j = 0; j < 8; ++j) {
                float vv = a[j] * alpha + bv;
                if (RELU) vv = fmaxf(vv, 0.0f);
                if constexpr (std::is_same<TC, _Float16>::value)
                    Cg[(long)(rbase + ti * 16 + j) * ldc + col] = (_Float16)vv;
                else
                    Cg[(long)(rbase + ti * 16 + j) * ldc + col] = vv;
            }
        }
    }
}

// ---------------- weight transpose+convert: [K,N] f32 -> [N,K] f16 ----------------
__global__ __launch_bounds__(256) void convert_transpose_kernel(
    const float* __restrict__ in, _Float16* __restrict__ out, int K, int N)
{
    const long mat = blockIdx.z;
    const float* pin = in + mat * (long)K * N;
    _Float16* pout = out + mat * (long)K * N;
    long i = (long)blockIdx.x * blockDim.x + threadIdx.x;
    if (i >= (long)K * N) return;
    int n = (int)(i % N);
    int kk = (int)(i / N);
    pout[(long)n * K + kk] = (_Float16)pin[i];
}

// ---------------- embedding (+optional sinusoidal PE) ----------------
template<bool POS>
__global__ __launch_bounds__(256) void embed_kernel(const int* __restrict__ tok,
                                                    const float* __restrict__ emb,
                                                    float* __restrict__ out)
{
    long i = (long)blockIdx.x * blockDim.x + threadIdx.x;   // < rows*DD
    int d = (int)(i % DD);
    long bs = i / DD;
    int s = (int)(bs % SS);
    int t = tok[bs];
    float v = emb[(long)t * DD + d];
    if (POS) {
        float freq = __powf(10000.0f, (2.0f * (float)(d >> 1)) / (float)DD);
        float ang  = (float)s / freq;
        v += (d & 1) ? __cosf(ang) : __sinf(ang);
    }
    out[i] = v;
}

// ---------------- residual + LayerNorm over D=512 (in-place safe) ----------------
__global__ __launch_bounds__(256) void ln_kernel(const float* __restrict__ a,
                                                 const float* __restrict__ x,
                                                 const float* __restrict__ g,
                                                 const float* __restrict__ b,
                                                 float* __restrict__ out)
{
    __shared__ float red[256];
    const long row = blockIdx.x;
    const int t = threadIdx.x;
    float v0 = x[row * DD + t];
    float v1 = x[row * DD + t + 256];
    if (a) { v0 += a[row * DD + t]; v1 += a[row * DD + t + 256]; }
    red[t] = v0 + v1; __syncthreads();
    for (int s = 128; s > 0; s >>= 1) { if (t < s) red[t] += red[t + s]; __syncthreads(); }
    float mean = red[0] * (1.0f / (float)DD);
    __syncthreads();
    float d0 = v0 - mean, d1 = v1 - mean;
    red[t] = d0 * d0 + d1 * d1; __syncthreads();
    for (int s = 128; s > 0; s >>= 1) { if (t < s) red[t] += red[t + s]; __syncthreads(); }
    float rstd = rsqrtf(red[0] * (1.0f / (float)DD) + 1e-5f);
    out[row * DD + t]       = d0 * rstd * g[t]       + b[t];
    out[row * DD + t + 256] = d1 * rstd * g[t + 256] + b[t + 256];
}

// ---------------- masked attention softmax over S=512, fp32 in -> f16 out ----------------
// row = ((b*H + h)*T + t) ; att already scaled by 1/sqrt(D) in the GEMM
__global__ __launch_bounds__(256) void attn_softmax_kernel(const float* __restrict__ att,
                                                           _Float16* __restrict__ out16,
                                                           const float* __restrict__ pad,
                                                           const float* __restrict__ subsq)
{
    __shared__ float red[256];
    const long row = blockIdx.x;
    const int t = (int)(row % TT);
    const int b = (int)(row / ((long)HH * TT));
    const int tid = threadIdx.x;
    const float* p = att + row * (long)SS;
    float v0 = p[tid], v1 = p[tid + 256];
    if (pad)   { v0 += pad[(long)b * SS + tid];   v1 += pad[(long)b * SS + tid + 256]; }
    if (subsq) { v0 += subsq[(long)t * SS + tid]; v1 += subsq[(long)t * SS + tid + 256]; }
    red[tid] = fmaxf(v0, v1); __syncthreads();
    for (int s = 128; s > 0; s >>= 1) { if (tid < s) red[tid] = fmaxf(red[tid], red[tid + s]); __syncthreads(); }
    float mx = red[0]; __syncthreads();
    float e0 = __expf(v0 - mx), e1 = __expf(v1 - mx);
    red[tid] = e0 + e1; __syncthreads();
    for (int s = 128; s > 0; s >>= 1) { if (tid < s) red[tid] += red[tid + s]; __syncthreads(); }
    float inv = 1.0f / red[0];
    _Float16* o = out16 + row * (long)SS;
    o[tid] = (_Float16)(e0 * inv); o[tid + 256] = (_Float16)(e1 * inv);
}

// ---------------- vocab softmax over 32000 (in-place on d_out) ----------------
__global__ __launch_bounds__(256) void vocab_softmax_kernel(float* __restrict__ logits)
{
    __shared__ float red[256];
    const long row = blockIdx.x;
    float* p = logits + row * (long)VOUT_;
    const int t = threadIdx.x;
    float mx = -3.0e38f;
    for (int i = t; i < VOUT_; i += 256) mx = fmaxf(mx, p[i]);
    red[t] = mx; __syncthreads();
    for (int s = 128; s > 0; s >>= 1) { if (t < s) red[t] = fmaxf(red[t], red[t + s]); __syncthreads(); }
    mx = red[0]; __syncthreads();
    float sum = 0.0f;
    for (int i = t; i < VOUT_; i += 256) sum += __expf(p[i] - mx);
    red[t] = sum; __syncthreads();
    for (int s = 128; s > 0; s >>= 1) { if (t < s) red[t] += red[t + s]; __syncthreads(); }
    float inv = 1.0f / red[0];
    for (int i = t; i < VOUT_; i += 256) p[i] = __expf(p[i] - mx) * inv;
}

// ---------------- copy head H-1 slice: [B,H,S,D] f16 -> [B,S,D] f16 ----------------
__global__ __launch_bounds__(256) void copy_head_kernel(const _Float16* __restrict__ kv,
                                                        _Float16* __restrict__ out)
{
    long i = (long)blockIdx.x * blockDim.x + threadIdx.x;  // < B*S*D
    int d = (int)(i % DD);
    long bs = i / DD;
    int s = (int)(bs % SS);
    int b = (int)(bs / SS);
    out[i] = kv[((((long)b * HH + (HH - 1)) * SS) + s) * (long)DD + d];
}

// ---------------- host-side launch helper ----------------
template<typename TA, typename TB, typename TC, bool TRANSB, bool RELU>
static inline void rungemm(hipStream_t st,
    const TA* A, long bA, long sA, int lda,
    const TB* B, long bB, long sB, int ldb,
    TC* C, long bC, long sC, int ldc,
    int M, int N, int K, int Z, const float* bias, float alpha)
{
    dim3 g(N / BN, M / BM, Z), blk(256);
    gemm_wmma<TA, TB, TRANSB, RELU, TC><<<g, blk, 0, st>>>(
        A, bA, sA, lda, B, bB, sB, ldb, C, bC, sC, ldc, M, N, K, bias, alpha);
}

typedef _Float16 h16;

extern "C" void kernel_launch(void* const* d_in, const int* in_sizes, int n_in,
                              void* d_out, int out_size, void* d_ws, size_t ws_size,
                              hipStream_t stream)
{
    (void)in_sizes; (void)n_in; (void)out_size; (void)ws_size;
    // ---- inputs (setup_inputs order) ----
    const int*   src        = (const int*)  d_in[0];
    const int*   tgt        = (const int*)  d_in[1];
    const float* src_pad    = (const float*)d_in[2];
    const float* src_subsq  = (const float*)d_in[3];
    const float* tgt_pad    = (const float*)d_in[4];
    const float* tgt_subsq  = (const float*)d_in[5];
    const float* input_emb  = (const float*)d_in[6];
    const float* outp_emb   = (const float*)d_in[7];
    const float* enc_attn_w = (const float*)d_in[8];
    const float* enc_attn_b = (const float*)d_in[9];
    const float* enc_proj_w = (const float*)d_in[10];
    const float* enc_proj_b = (const float*)d_in[11];
    const float* enc_ln1_g  = (const float*)d_in[12];
    const float* enc_ln1_b  = (const float*)d_in[13];
    const float* enc_ff1_w  = (const float*)d_in[14];
    const float* enc_ff1_b  = (const float*)d_in[15];
    const float* enc_ff2_w  = (const float*)d_in[16];
    const float* enc_ff2_b  = (const float*)d_in[17];
    const float* enc_ln2_g  = (const float*)d_in[18];
    const float* enc_ln2_b  = (const float*)d_in[19];
    const float* dec_self_w = (const float*)d_in[20];
    const float* dec_self_b = (const float*)d_in[21];
    const float* dec_self_pw= (const float*)d_in[22];
    const float* dec_self_pb= (const float*)d_in[23];
    const float* dec_ln1_g  = (const float*)d_in[24];
    const float* dec_ln1_b  = (const float*)d_in[25];
    const float* dec_memq_w = (const float*)d_in[26];
    const float* dec_memq_b = (const float*)d_in[27];
    const float* dec_mem_pw = (const float*)d_in[28];
    const float* dec_mem_pb = (const float*)d_in[29];
    const float* dec_ln2_g  = (const float*)d_in[30];
    const float* dec_ln2_b  = (const float*)d_in[31];
    const float* dec_ff1_w  = (const float*)d_in[32];
    const float* dec_ff1_b  = (const float*)d_in[33];
    const float* dec_ff2_w  = (const float*)d_in[34];
    const float* dec_ff2_b  = (const float*)d_in[35];
    const float* dec_ln3_g  = (const float*)d_in[36];
    const float* dec_ln3_b  = (const float*)d_in[37];
    const float* dec_norm_g = (const float*)d_in[38];
    const float* dec_norm_b = (const float*)d_in[39];
    const float* out_w      = (const float*)d_in[40];
    const float* out_b      = (const float*)d_in[41];
    float* logits = (float*)d_out;

    // ---- workspace partition ----
    char* wsb = (char*)d_ws;
    auto alloc_f32 = [&](long n) { float* p = (float*)wsb; wsb += n * sizeof(float); return p; };
    auto alloc_f16 = [&](long n) { h16* p = (h16*)wsb; wsb += n * sizeof(h16); return p; };

    const long N_X   = (long)BB * SS * DD;            // 1M
    const long N_QKV = (long)BB * HH * TT * DD;       // 8M
    const long N_ATT = (long)BB * HH * TT * SS;       // 8M
    const long N_CAT = (long)BB * TT * HH * DD;       // 8M
    const long N_FF  = (long)BB * TT * FF_;           // 4M

    float* x    = alloc_f32(N_X);
    float* y    = alloc_f32(N_X);
    float* abuf = alloc_f32(N_X);
    float* att  = alloc_f32(N_ATT);        // raw scores (fp32 for softmax)
    h16* q16    = alloc_f16(N_QKV);
    h16* k16    = alloc_f16(N_QKV);
    h16* v16    = alloc_f16(N_QKV);
    h16* p16    = alloc_f16(N_ATT);        // softmax probabilities
    h16* cat16  = alloc_f16(N_CAT);
    h16* ffh16  = alloc_f16(N_FF);
    h16* memk16 = alloc_f16(N_X);
    h16* memv16 = alloc_f16(N_X);
    // f16 transposed weights
    h16* w_enc_attn = alloc_f16((long)LL * 3 * HH * DD * DD);
    h16* w_dec_self = alloc_f16((long)LL * 3 * HH * DD * DD);
    h16* w_memq     = alloc_f16((long)LL * HH * DD * DD);
    h16* w_enc_proj = alloc_f16((long)LL * HH * DD * DD);
    h16* w_dec_sprj = alloc_f16((long)LL * HH * DD * DD);
    h16* w_dec_mprj = alloc_f16((long)LL * HH * DD * DD);
    h16* w_enc_ff1  = alloc_f16((long)LL * DD * FF_);
    h16* w_enc_ff2  = alloc_f16((long)LL * FF_ * DD);
    h16* w_dec_ff1  = alloc_f16((long)LL * DD * FF_);
    h16* w_dec_ff2  = alloc_f16((long)LL * FF_ * DD);
    h16* w_out      = alloc_f16((long)DD * VOUT_);

    const float inv_sqrt_d = 0.044194173824159216f;   // 1/sqrt(512)
    const long rows = (long)BB * SS;                  // 2048 (== B*T)
    const long DxD = (long)DD * DD;

    // ---- one-time weight transpose+convert to f16 [N][K] ----
    auto conv = [&](const float* in, h16* out, int K, int N, int nmat) {
        long per = (long)K * N;
        dim3 g((unsigned)((per + 255) / 256), 1, nmat);
        convert_transpose_kernel<<<g, 256, 0, stream>>>(in, out, K, N);
    };
    conv(enc_attn_w,  w_enc_attn, DD, DD, LL * 3 * HH);
    conv(dec_self_w,  w_dec_self, DD, DD, LL * 3 * HH);
    conv(dec_memq_w,  w_memq,     DD, DD, LL * HH);
    conv(enc_proj_w,  w_enc_proj, HH * DD, DD, LL);
    conv(dec_self_pw, w_dec_sprj, HH * DD, DD, LL);
    conv(dec_mem_pw,  w_dec_mprj, HH * DD, DD, LL);
    conv(enc_ff1_w,   w_enc_ff1,  DD, FF_, LL);
    conv(enc_ff2_w,   w_enc_ff2,  FF_, DD, LL);
    conv(dec_ff1_w,   w_dec_ff1,  DD, FF_, LL);
    conv(dec_ff2_w,   w_dec_ff2,  FF_, DD, LL);
    conv(out_w,       w_out,      DD, VOUT_, 1);

    // ================= encoder =================
    embed_kernel<true><<<(unsigned)(N_X / 256), 256, 0, stream>>>(src, input_emb, x);

    for (int l = 0; l < LL; ++l) {
        // QKV: per (matrix m, head h) GEMM, batched over B; weights f16 [N][K]
        for (int m = 0; m < 3; ++m) {
            h16* dst = (m == 0) ? k16 : (m == 1) ? v16 : q16;
            for (int h = 0; h < HH; ++h) {
                const h16* W = w_enc_attn + ((((long)l * 3 + m) * HH + h) * DxD);
                const float* Bv = enc_attn_b + (((long)l * 3 + m) * HH + h) * DD;
                rungemm<float, h16, h16, true, false>(stream,
                        x, 0, (long)SS * DD, DD,
                        W, 0, 0, DD,
                        dst, (long)h * SS * DD, (long)HH * SS * DD, DD,
                        SS, DD, DD, BB, Bv, 1.0f);
            }
        }
        // scores = q k^T / sqrt(D), batched over B*H (all-f16 async path)
        rungemm<h16, h16, float, true, false>(stream,
                q16, 0, (long)SS * DD, DD,
                k16, 0, (long)SS * DD, DD,
                att, 0, (long)SS * SS, SS,
                SS, SS, DD, BB * HH, nullptr, inv_sqrt_d);
        attn_softmax_kernel<<<(unsigned)((long)BB * HH * SS), 256, 0, stream>>>(att, p16, src_pad, src_subsq);
        // context -> cat[b,t,h*D+e], per head (A f16 async, B f16 NN)
        for (int h = 0; h < HH; ++h) {
            rungemm<h16, h16, h16, false, false>(stream,
                    p16, (long)h * SS * SS, (long)HH * SS * SS, SS,
                    v16, (long)h * SS * DD, (long)HH * SS * DD, DD,
                    cat16, (long)h * DD, (long)SS * HH * DD, HH * DD,
                    SS, DD, SS, BB, nullptr, 1.0f);
        }
        // proj + residual LN
        rungemm<h16, h16, float, true, false>(stream,
                cat16, 0, 0, HH * DD,
                w_enc_proj + (long)l * HH * DD * DD, 0, 0, HH * DD,
                abuf, 0, 0, DD,
                (int)rows, DD, HH * DD, 1, enc_proj_b + (long)l * DD, 1.0f);
        ln_kernel<<<(unsigned)rows, 256, 0, stream>>>(abuf, x, enc_ln1_g + (long)l * DD, enc_ln1_b + (long)l * DD, x);
        // FF
        rungemm<float, h16, h16, true, true>(stream,
                x, 0, 0, DD,
                w_enc_ff1 + (long)l * DD * FF_, 0, 0, DD,
                ffh16, 0, 0, FF_,
                (int)rows, FF_, DD, 1, enc_ff1_b + (long)l * FF_, 1.0f);
        rungemm<h16, h16, float, true, false>(stream,
                ffh16, 0, 0, FF_,
                w_enc_ff2 + (long)l * FF_ * DD, 0, 0, FF_,
                abuf, 0, 0, DD,
                (int)rows, DD, FF_, 1, enc_ff2_b + (long)l * DD, 1.0f);
        ln_kernel<<<(unsigned)rows, 256, 0, stream>>>(abuf, x, enc_ln2_g + (long)l * DD, enc_ln2_b + (long)l * DD, x);
    }
    // memory keys/values = last encoder layer, head H-1
    copy_head_kernel<<<(unsigned)(N_X / 256), 256, 0, stream>>>(k16, memk16);
    copy_head_kernel<<<(unsigned)(N_X / 256), 256, 0, stream>>>(v16, memv16);

    // ================= decoder =================
    embed_kernel<false><<<(unsigned)(N_X / 256), 256, 0, stream>>>(tgt, outp_emb, y);

    for (int l = 0; l < LL; ++l) {
        // --- masked self-attention ---
        for (int m = 0; m < 3; ++m) {
            h16* dst = (m == 0) ? k16 : (m == 1) ? v16 : q16;
            for (int h = 0; h < HH; ++h) {
                const h16* W = w_dec_self + ((((long)l * 3 + m) * HH + h) * DxD);
                const float* Bv = dec_self_b + (((long)l * 3 + m) * HH + h) * DD;
                rungemm<float, h16, h16, true, false>(stream,
                        y, 0, (long)TT * DD, DD,
                        W, 0, 0, DD,
                        dst, (long)h * TT * DD, (long)HH * TT * DD, DD,
                        TT, DD, DD, BB, Bv, 1.0f);
            }
        }
        rungemm<h16, h16, float, true, false>(stream,
                q16, 0, (long)TT * DD, DD,
                k16, 0, (long)TT * DD, DD,
                att, 0, (long)TT * TT, TT,
                TT, TT, DD, BB * HH, nullptr, inv_sqrt_d);
        attn_softmax_kernel<<<(unsigned)((long)BB * HH * TT), 256, 0, stream>>>(att, p16, tgt_pad, tgt_subsq);
        for (int h = 0; h < HH; ++h) {
            rungemm<h16, h16, h16, false, false>(stream,
                    p16, (long)h * TT * TT, (long)HH * TT * TT, TT,
                    v16, (long)h * TT * DD, (long)HH * TT * DD, DD,
                    cat16, (long)h * DD, (long)TT * HH * DD, HH * DD,
                    TT, DD, TT, BB, nullptr, 1.0f);
        }
        rungemm<h16, h16, float, true, false>(stream,
                cat16, 0, 0, HH * DD,
                w_dec_sprj + (long)l * HH * DD * DD, 0, 0, HH * DD,
                abuf, 0, 0, DD,
                (int)rows, DD, HH * DD, 1, dec_self_pb + (long)l * DD, 1.0f);
        ln_kernel<<<(unsigned)rows, 256, 0, stream>>>(abuf, y, dec_ln1_g + (long)l * DD, dec_ln1_b + (long)l * DD, y);

        // --- memory (cross) attention ---
        for (int h = 0; h < HH; ++h) {
            const h16* W = w_memq + (((long)l * HH + h) * DxD);
            const float* Bv = dec_memq_b + ((long)l * HH + h) * DD;
            rungemm<float, h16, h16, true, false>(stream,
                    y, 0, (long)TT * DD, DD,
                    W, 0, 0, DD,
                    q16, (long)h * TT * DD, (long)HH * TT * DD, DD,
                    TT, DD, DD, BB, Bv, 1.0f);
        }
        for (int h = 0; h < HH; ++h) {
            rungemm<h16, h16, float, true, false>(stream,
                    q16, (long)h * TT * DD, (long)HH * TT * DD, DD,
                    memk16, 0, (long)SS * DD, DD,
                    att, (long)h * TT * SS, (long)HH * TT * SS, SS,
                    TT, SS, DD, BB, nullptr, inv_sqrt_d);
        }
        attn_softmax_kernel<<<(unsigned)((long)BB * HH * TT), 256, 0, stream>>>(att, p16, src_pad, nullptr);
        for (int h = 0; h < HH; ++h) {
            rungemm<h16, h16, h16, false, false>(stream,
                    p16, (long)h * TT * SS, (long)HH * TT * SS, SS,
                    memv16, 0, (long)SS * DD, DD,
                    cat16, (long)h * DD, (long)TT * HH * DD, HH * DD,
                    TT, DD, SS, BB, nullptr, 1.0f);
        }
        rungemm<h16, h16, float, true, false>(stream,
                cat16, 0, 0, HH * DD,
                w_dec_mprj + (long)l * HH * DD * DD, 0, 0, HH * DD,
                abuf, 0, 0, DD,
                (int)rows, DD, HH * DD, 1, dec_mem_pb + (long)l * DD, 1.0f);
        ln_kernel<<<(unsigned)rows, 256, 0, stream>>>(abuf, y, dec_ln2_g + (long)l * DD, dec_ln2_b + (long)l * DD, y);

        // --- FF ---
        rungemm<float, h16, h16, true, true>(stream,
                y, 0, 0, DD,
                w_dec_ff1 + (long)l * DD * FF_, 0, 0, DD,
                ffh16, 0, 0, FF_,
                (int)rows, FF_, DD, 1, dec_ff1_b + (long)l * FF_, 1.0f);
        rungemm<h16, h16, float, true, false>(stream,
                ffh16, 0, 0, FF_,
                w_dec_ff2 + (long)l * FF_ * DD, 0, 0, FF_,
                abuf, 0, 0, DD,
                (int)rows, DD, FF_, 1, dec_ff2_b + (long)l * DD, 1.0f);
        ln_kernel<<<(unsigned)rows, 256, 0, stream>>>(abuf, y, dec_ln3_g + (long)l * DD, dec_ln3_b + (long)l * DD, y);
    }

    // final LN (no residual)
    ln_kernel<<<(unsigned)rows, 256, 0, stream>>>(nullptr, y, dec_norm_g, dec_norm_b, y);

    // logits = y @ out_w^T + out_b ; then row softmax over VOUT
    rungemm<float, h16, float, true, false>(stream,
            y, 0, 0, DD,
            w_out, 0, 0, DD,
            logits, 0, 0, VOUT_,
            (int)rows, VOUT_, DD, 1, out_b, 1.0f);
    vocab_softmax_kernel<<<(unsigned)rows, 256, 0, stream>>>(logits);
}
